// TransformerDecoder_64510408786213
// MI455X (gfx1250) — matrix operand order repaired and checked by reading, not compiled, not run on hardware
//
#include <hip/hip_runtime.h>

typedef __attribute__((ext_vector_type(16))) __bf16 bf16x16;
typedef __attribute__((ext_vector_type(8)))  float  f32x8;

#define ACT_NONE 0
#define ACT_RELU 1
#define ACT_SOFTPLUS 2

struct ConvDesc {
  int C, Cp, T;                                  // in-channels, padded, taps
  int KD, KH, KW, SD, SH, SW, PD, PH, PW, LD;    // LD = lhs_dilation (1|2)
  int Din, Hin, Win, Dout, Hout, Wout;
};

__device__ __forceinline__ void cvt16(bf16x16& va, const float4& a0, const float4& a1,
                                      const float4& a2, const float4& a3) {
  va[0]=(__bf16)a0.x; va[1]=(__bf16)a0.y; va[2]=(__bf16)a0.z; va[3]=(__bf16)a0.w;
  va[4]=(__bf16)a1.x; va[5]=(__bf16)a1.y; va[6]=(__bf16)a1.z; va[7]=(__bf16)a1.w;
  va[8]=(__bf16)a2.x; va[9]=(__bf16)a2.y; va[10]=(__bf16)a2.z; va[11]=(__bf16)a2.w;
  va[12]=(__bf16)a3.x; va[13]=(__bf16)a3.y; va[14]=(__bf16)a3.z; va[15]=(__bf16)a3.w;
}
// A-fragment: two aligned 8-float runs per lane (rows of a 16x32 bf16 tile)
__device__ __forceinline__ void loadA(bf16x16& va, const float* arow, int k0, int half) {
  const float4* p = (const float4*)(arow + k0 + half * 8);
  const float4* q = (const float4*)(arow + k0 + 16 + half * 8);
  cvt16(va, p[0], p[1], q[0], q[1]);
}
// B-fragment: two aligned 8-float runs per lane (col of a 32x16 bf16 tile)
__device__ __forceinline__ void loadB(bf16x16& vb, const float* brow, int k0, int half) {
  const float4* bp = (const float4*)(brow + k0 + half * 8);
  const float4* bq = (const float4*)(brow + k0 + 16 + half * 8);
  cvt16(vb, bp[0], bp[1], bq[0], bq[1]);
}
__device__ __forceinline__ float epilog(float y, const float* bias, int ncol, int act) {
  if (bias) y += bias[ncol];
  if (act == ACT_RELU)          y = fmaxf(y, 0.0f);
  else if (act == ACT_SOFTPLUS) y = (y > 20.0f) ? y : __logf(1.0f + __expf(y));
  return y;
}

// ---------------------------------------------------------------------------
// Conv GEMM (implicit im2col, tap-major K). One wave = 32x16 output tile:
// two A fragments share one B fragment -> 2 WMMAs per B load. All loads are
// unconditional aligned float4 (volume stride Cp, weights pre-padded).
// ---------------------------------------------------------------------------
__global__ void k_gemm_conv(const float* __restrict__ Avol, const float* __restrict__ Bw,
                            const float* __restrict__ bias, const float* __restrict__ addsrc,
                            float* __restrict__ Cout,
                            int M, int N, int ldc, int act, ConvDesc cd) {
  int wave  = threadIdx.x >> 5;
  int lane  = threadIdx.x & 31;
  int tileN = blockIdx.x;
  int tmm   = blockIdx.y * 8 + wave;           // 32-row super-tile
  if (tmm * 32 >= M) return;                   // wave-uniform
  int half  = lane >> 4;
  int mrow0 = tmm * 32 + (lane & 15);
  int mrow1 = mrow0 + 16;
  int ncol  = tileN * 16 + (lane & 15);

  // decode both output voxels once
  int spat = cd.Dout * cd.Hout * cd.Wout;
  int hw   = cd.Hout * cd.Wout;
  int b0 = mrow0 / spat, r0 = mrow0 - b0 * spat;
  int od0 = r0 / hw; r0 -= od0 * hw;
  int oh0 = r0 / cd.Wout, ow0 = r0 - oh0 * cd.Wout;
  int b1 = mrow1 / spat, r1 = mrow1 - b1 * spat;
  int od1 = r1 / hw; r1 -= od1 * hw;
  int oh1 = r1 / cd.Wout, ow1 = r1 - oh1 * cd.Wout;

  const float* brow = Bw + (long)ncol * ((long)cd.T * cd.Cp);
  f32x8 acc0 = {0.f,0.f,0.f,0.f,0.f,0.f,0.f,0.f};
  f32x8 acc1 = {0.f,0.f,0.f,0.f,0.f,0.f,0.f,0.f};

  int kd = 0, kh = 0, kw = 0;
  for (int t = 0; t < cd.T; ++t) {
    int gd0 = od0 * cd.SD - cd.PD + kd, gh0 = oh0 * cd.SH - cd.PH + kh, gw0 = ow0 * cd.SW - cd.PW + kw;
    int gd1 = od1 * cd.SD - cd.PD + kd, gh1 = oh1 * cd.SH - cd.PH + kh, gw1 = ow1 * cd.SW - cd.PW + kw;
    bool v0 = true, v1 = true;
    if (cd.LD == 2) {                          // lhs-dilated (ConvTranspose)
      if ((gd0 | gh0 | gw0) & 1) v0 = false;
      if ((gd1 | gh1 | gw1) & 1) v1 = false;
      gd0 >>= 1; gh0 >>= 1; gw0 >>= 1; gd1 >>= 1; gh1 >>= 1; gw1 >>= 1;
    }
    v0 = v0 && gd0 >= 0 && gd0 < cd.Din && gh0 >= 0 && gh0 < cd.Hin && gw0 >= 0 && gw0 < cd.Win;
    v1 = v1 && gd1 >= 0 && gd1 < cd.Din && gh1 >= 0 && gh1 < cd.Hin && gw1 >= 0 && gw1 < cd.Win;
    const float* ar0 = Avol + ((((long)b0 * cd.Din + gd0) * cd.Hin + gh0) * cd.Win + gw0) * cd.Cp;
    const float* ar1 = Avol + ((((long)b1 * cd.Din + gd1) * cd.Hin + gh1) * cd.Win + gw1) * cd.Cp;
    const float* bt  = brow + (long)t * cd.Cp;

    for (int c0 = 0; c0 < cd.Cp; c0 += 32) {
      bf16x16 vb, va0, va1;
      loadB(vb, bt, c0, half);
      float4 z = {0,0,0,0};
      if (v0) loadA(va0, ar0, c0, half); else cvt16(va0, z, z, z, z);
      if (v1) loadA(va1, ar1, c0, half); else cvt16(va1, z, z, z, z);
      acc0 = __builtin_amdgcn_wmma_f32_16x16x32_bf16(false, va0, false, vb, (short)0, acc0, false, false);
      acc1 = __builtin_amdgcn_wmma_f32_16x16x32_bf16(false, va1, false, vb, (short)0, acc1, false, false);
    }
    if (++kw == cd.KW) { kw = 0; if (++kh == cd.KH) { kh = 0; ++kd; } }
  }

#pragma unroll
  for (int v = 0; v < 8; ++v) {
    int mg0 = tmm * 32 + v + 8 * half;
    int mg1 = mg0 + 16;
    float y0 = (ncol < N) ? epilog(acc0[v], bias, ncol, act) : 0.0f;
    float y1 = (ncol < N) ? epilog(acc1[v], bias, ncol, act) : 0.0f;
    if (addsrc) { y0 += addsrc[(long)mg0 * ldc + ncol];     // pad cols hold 0
                  y1 += addsrc[(long)mg1 * ldc + ncol]; }
    Cout[(long)mg0 * ldc + ncol] = y0;                      // keep pad zeroed
    Cout[(long)mg1 * ldc + ncol] = y1;
  }
}

// ---------------------------------------------------------------------------
// Direct GEMM: A[M,Kp] (padded stride, zero pad) * B[Np,Kp] (pre-padded).
// 32x16 super-tile per wave; fully unconditional aligned float4 loads.
// ---------------------------------------------------------------------------
__global__ void k_gemm_dir(const float* __restrict__ A, const float* __restrict__ Bw,
                           const float* __restrict__ bias, float* __restrict__ Cout,
                           int M, int N, int Kp, int lda, int ldc, int act, int padstore) {
  int wave  = threadIdx.x >> 5;
  int lane  = threadIdx.x & 31;
  int tileN = blockIdx.x;
  int tmm   = blockIdx.y * 8 + wave;
  if (tmm * 32 >= M) return;
  int half  = lane >> 4;
  int mrow0 = tmm * 32 + (lane & 15);
  int ncol  = tileN * 16 + (lane & 15);

  const float* arow0 = A + (long)mrow0 * lda;
  const float* arow1 = arow0 + (long)16 * lda;
  const float* brow  = Bw + (long)ncol * Kp;

  f32x8 acc0 = {0.f,0.f,0.f,0.f,0.f,0.f,0.f,0.f};
  f32x8 acc1 = {0.f,0.f,0.f,0.f,0.f,0.f,0.f,0.f};
  for (int k0 = 0; k0 < Kp; k0 += 32) {
    bf16x16 vb, va0, va1;
    loadB(vb, brow, k0, half);
    loadA(va0, arow0, k0, half);
    loadA(va1, arow1, k0, half);
    acc0 = __builtin_amdgcn_wmma_f32_16x16x32_bf16(false, va0, false, vb, (short)0, acc0, false, false);
    acc1 = __builtin_amdgcn_wmma_f32_16x16x32_bf16(false, va1, false, vb, (short)0, acc1, false, false);
  }
#pragma unroll
  for (int v = 0; v < 8; ++v) {
    int mg0 = tmm * 32 + v + 8 * half;
    int mg1 = mg0 + 16;
    float y0 = (ncol < N) ? epilog(acc0[v], bias, ncol, act) : 0.0f;
    float y1 = (ncol < N) ? epilog(acc1[v], bias, ncol, act) : 0.0f;
    if (padstore) {
      Cout[(long)mg0 * ldc + ncol] = y0;
      Cout[(long)mg1 * ldc + ncol] = y1;
    } else if (ncol < N) {
      Cout[(long)mg0 * ldc + ncol] = y0;
      Cout[(long)mg1 * ldc + ncol] = y1;
    }
  }
}

// NCDHW -> channels-last rows [B*S, Cp], zero-padded columns
__global__ void k_nc2rc(const float* __restrict__ in, float* __restrict__ out,
                        int B, int C, int Cp, int S) {
  long i = (long)blockIdx.x * blockDim.x + threadIdx.x;
  long total = (long)B * S * Cp;
  if (i >= total) return;
  int c = (int)(i % Cp);
  long rr = i / Cp;
  int v = (int)(rr % S);
  int b = (int)(rr / S);
  out[i] = (c < C) ? in[((long)b * C + c) * S + v] : 0.0f;
}

// linear weight w[N][K] -> out[Np][Kp] zero padded
__global__ void k_repack_lin(const float* __restrict__ w, float* __restrict__ out,
                             int N, int K, int Np, int Kp) {
  long i = (long)blockIdx.x * blockDim.x + threadIdx.x;
  if (i >= (long)Np * Kp) return;
  int k = (int)(i % Kp);
  int o = (int)(i / Kp);
  out[i] = (o < N && k < K) ? w[(long)o * K + k] : 0.0f;
}

// conv weight w[N][C][T] -> out[Np][T][Cp] (tap-major, zero padded)
__global__ void k_repack_conv(const float* __restrict__ w, float* __restrict__ out,
                              int N, int C, int T, int Np, int Cp) {
  long i = (long)blockIdx.x * blockDim.x + threadIdx.x;
  if (i >= (long)Np * T * Cp) return;
  int ci = (int)(i % Cp);
  int t  = (int)((i / Cp) % T);
  int o  = (int)(i / ((long)Cp * T));
  out[i] = (o < N && ci < C) ? w[((long)o * C + ci) * T + t] : 0.0f;
}

// ConvTranspose weight w[Cin][Cout][64] -> out[Np][64][Cp], flip+transpose
__global__ void k_repack_ct(const float* __restrict__ w, float* __restrict__ out,
                            int Cin, int Cout, int Np, int Cp) {
  long i = (long)blockIdx.x * blockDim.x + threadIdx.x;
  if (i >= (long)Np * 64 * Cp) return;
  int ci = (int)(i % Cp);
  int t  = (int)((i / Cp) % 64);
  int o  = (int)(i / ((long)Cp * 64));
  out[i] = (o < Cout && ci < Cin) ? w[((long)ci * Cout + o) * 64 + (63 - t)] : 0.0f;
}

// Per-channel mean/var over rows (population) — one block per channel
__global__ void k_bn_stats(const float* __restrict__ x, float* __restrict__ stats,
                           int Mrows, int C, int Cp) {
  int c = blockIdx.x;
  float s = 0.f, s2 = 0.f;
  for (int r = threadIdx.x; r < Mrows; r += blockDim.x) {
    float v = x[(long)r * Cp + c];
    s += v; s2 += v * v;
  }
  __shared__ float sh0[256];
  __shared__ float sh1[256];
  sh0[threadIdx.x] = s; sh1[threadIdx.x] = s2;
  __syncthreads();
  for (int st = 128; st > 0; st >>= 1) {
    if (threadIdx.x < st) { sh0[threadIdx.x] += sh0[threadIdx.x + st];
                            sh1[threadIdx.x] += sh1[threadIdx.x + st]; }
    __syncthreads();
  }
  if (threadIdx.x == 0) {
    float m = sh0[0] / (float)Mrows;
    stats[c]     = m;
    stats[C + c] = sh1[0] / (float)Mrows - m * m;
  }
}

__global__ void k_bn_relu(const float* __restrict__ x, const float* __restrict__ stats,
                          const float* __restrict__ g, const float* __restrict__ be,
                          float* __restrict__ y, long total, int C, int Cp) {
  long i = (long)blockIdx.x * blockDim.x + threadIdx.x;
  if (i >= total) return;
  int c = (int)(i % Cp);
  float r = 0.0f;
  if (c < C) {
    float m = stats[c], v = stats[C + c];
    r = fmaxf((x[i] - m) * rsqrtf(v + 1e-5f) * g[c] + be[c], 0.0f);
  }
  y[i] = r;
}

// Causal depthwise conv1d (k=4, left pad 3) + SiLU.  U,UC: [B*L, dip]
__global__ void k_dwconv_silu(const float* __restrict__ U, const float* __restrict__ cw,
                              const float* __restrict__ cb, float* __restrict__ UC,
                              int Bn, int L, int di, int dip) {
  long i = (long)blockIdx.x * blockDim.x + threadIdx.x;
  if (i >= (long)Bn * L * dip) return;
  int ch = (int)(i % dip);
  long rr = i / dip;
  int l = (int)(rr % L);
  int b = (int)(rr / L);
  float out = 0.0f;
  if (ch < di) {
    float acc = cb[ch];
#pragma unroll
    for (int j = 0; j < 4; ++j) {
      int ls = l - 3 + j;
      if (ls >= 0) acc += U[((long)b * L + ls) * dip + ch] * cw[ch * 4 + j];
    }
    out = acc / (1.0f + __expf(-acc));     // SiLU
  }
  UC[i] = out;
}

// Selective scan: thread=(b,d,n); 16 lanes share (b,d), reduce via shfl_xor.
__global__ void k_scan(const float* __restrict__ DT, const float* __restrict__ UC,
                       const float* __restrict__ XD, const float* __restrict__ Zb,
                       const float* __restrict__ Alog, const float* __restrict__ Dp,
                       float* __restrict__ SCO,
                       int Bn, int L, int di, int dip, int dtr, int nxp) {
  int t = blockIdx.x * blockDim.x + threadIdx.x;
  if (t >= Bn * di * 16) return;
  int n = t & 15;
  int d = (t >> 4) % di;
  int b = t / (16 * di);
  float Aval = -__expf(Alog[d * 16 + n]);
  float Dval = Dp[d];
  float h = 0.0f;
  for (int l = 0; l < L; ++l) {
    long row = (long)b * L + l;
    float dt = DT[row * dip + d];
    float u  = UC[row * dip + d];
    float Bv = XD[row * nxp + dtr + n];
    float Cv = XD[row * nxp + dtr + 16 + n];
    h = h * __expf(dt * Aval) + dt * u * Bv;
    float p = h * Cv;
    p += __shfl_xor(p, 1, 32);
    p += __shfl_xor(p, 2, 32);
    p += __shfl_xor(p, 4, 32);
    p += __shfl_xor(p, 8, 32);
    if (n == 0) {
      float y = p + u * Dval;
      float z = Zb[row * dip + d];
      SCO[row * dip + d] = y * (z / (1.0f + __expf(-z)));
    }
  }
}

// ---------------------------------------------------------------------------
static inline int p32(int x) { return (x + 31) & ~31; }

extern "C" void kernel_launch(void* const* d_in, const int* in_sizes, int n_in,
                              void* d_out, int out_size, void* d_ws, size_t ws_size,
                              hipStream_t stream) {
  (void)in_sizes; (void)n_in; (void)out_size; (void)ws_size;
  const int Bn = 2;
  auto F = [&](int i) { return (const float*)d_in[i]; };

  size_t off = 0;
  auto alloc = [&](size_t elems) {
    float* p = (float*)((char*)d_ws + off);
    off += ((elems * sizeof(float) + 255) / 256) * 256;
    return p;
  };
  float* ACTA  = alloc(4194304);   // max 65536*64
  float* ACTB  = alloc(4194304);
  float* R1    = alloc(53248);     // 128*416 resblock temps
  float* R2    = alloc(53248);
  float* Zb    = alloc(8388608);   // 65536*128
  float* Ub    = alloc(8388608);   // u; reused as scan output
  float* UCb   = alloc(8388608);
  float* DTb   = alloc(8388608);
  float* XDb   = alloc(4194304);   // 65536*64
  float* WREP  = alloc(6000000);   // repacked weights (max 224*64*416)
  float* STATS = alloc(512);

  // direct GEMM with weight pre-pad: W[N,K] -> WREP[Np,Kp], then 32x16 tiles
  auto lin = [&](const float* A, const float* W, const float* bias, float* C,
                 int M, int N, int K, int lda, int ldc, int act, int padstore) {
    int Kp = p32(K);
    int Np = padstore ? ldc : ((N + 15) & ~15);
    long tot = (long)Np * Kp;
    hipLaunchKernelGGL(k_repack_lin, dim3((int)((tot + 255) / 256)), dim3(256), 0, stream,
                       W, WREP, N, K, Np, Kp);
    dim3 grd(padstore ? (ldc / 16) : ((N + 15) / 16), (M / 32 + 7) / 8);
    hipLaunchKernelGGL(k_gemm_dir, grd, dim3(256), 0, stream,
                       A, WREP, bias, C, M, N, Kp, lda, ldc, act, padstore);
  };
  auto conv = [&](const float* A, const float* bias, const float* add, float* C,
                  int M, int N, int ldc, int act, ConvDesc cd) {
    dim3 grd(ldc / 16, (M / 32 + 7) / 8);
    hipLaunchKernelGGL(k_gemm_conv, grd, dim3(256), 0, stream,
                       A, WREP, bias, add, C, M, N, ldc, act, cd);
  };

  // ---- input NCDHW -> channels-last rows [128, 608], then 1x1x1 conv ----
  {
    long total = (long)Bn * 64 * 608;
    hipLaunchKernelGGL(k_nc2rc, dim3((int)((total + 255) / 256)), dim3(256), 0, stream,
                       F(0), ACTA, Bn, 588, 608, 64);
  }
  lin(ACTA, F(1), F(2), ACTB, Bn * 64, 392, 588, 608, 416, ACT_NONE, 1);
  float* cur = ACTB;   // [128, 416]

  // ---- 3 resblocks @392ch, 4^3 ----
  ConvDesc c1d = {392, 416, 16, 1,4,4, 1,2,2, 0,3,3, 1, 4,4,4, 4,4,4};
  ConvDesc c2d = {392, 416, 3,  3,1,1, 1,1,1, 1,0,0, 1, 4,4,4, 4,4,4};
  auto rp_conv = [&](int wi, int T) {
    long tot = (long)416 * T * 416;
    hipLaunchKernelGGL(k_repack_conv, dim3((int)((tot + 255) / 256)), dim3(256), 0, stream,
                       F(wi), WREP, 392, 392, T, 416, 416);
  };
  for (int r = 0; r < 3; ++r) {
    int base = 3 + r * 6;
    rp_conv(base+0, 16);
    conv(cur, nullptr, nullptr, R1, Bn*64, 392, 416, ACT_RELU, c1d);
    rp_conv(base+1, 3);
    conv(R1, nullptr, nullptr, R2, Bn*64, 392, 416, ACT_RELU, c2d);
    rp_conv(base+2, 16);
    conv(R2, nullptr, nullptr, R1, Bn*64, 392, 416, ACT_RELU, c1d);
    rp_conv(base+3, 3);
    conv(R1, nullptr, nullptr, R2, Bn*64, 392, 416, ACT_RELU, c2d);
    rp_conv(base+4, 16);
    conv(R2, nullptr, nullptr, R1, Bn*64, 392, 416, ACT_RELU, c1d);
    rp_conv(base+5, 3);
    conv(R1, nullptr, cur, cur, Bn*64, 392, 416, ACT_NONE, c2d);
  }

  // ---- ConvTranspose(k4,s2,p1) as lhs-dilated conv + BN + ReLU ----
  auto up_stage = [&](int wi, int bi, int gi, int bei, int Cin, int Cout, int Sin) {
    int Cp = p32(Cin), Np = p32(Cout);
    long tot = (long)Np * 64 * Cp;
    hipLaunchKernelGGL(k_repack_ct, dim3((int)((tot + 255) / 256)), dim3(256), 0, stream,
                       F(wi), WREP, Cin, Cout, Np, Cp);
    int Sout = Sin * 2;
    ConvDesc cd = {Cin, Cp, 64, 4,4,4, 1,1,1, 2,2,2, 2, Sin,Sin,Sin, Sout,Sout,Sout};
    int M = Bn * Sout * Sout * Sout;
    float* dst = (cur == ACTA) ? ACTB : ACTA;
    conv(cur, F(bi), nullptr, dst, M, Cout, Np, ACT_NONE, cd);
    hipLaunchKernelGGL(k_bn_stats, dim3(Cout), dim3(256), 0, stream, dst, STATS, M, Cout, Np);
    long total = (long)M * Np;
    hipLaunchKernelGGL(k_bn_relu, dim3((int)((total + 255) / 256)), dim3(256), 0, stream,
                       dst, STATS, F(gi), F(bei), dst, total, Cout, Np);
    cur = dst;
  };

  // ---- Mamba block (channels-last == seq layout) ----
  auto mamba_stage = [&](int base, int D, int L) {
    int di = 2 * D, Dp = p32(D), dip = p32(di);
    int dtr = (D + 15) / 16, nxp = 64;
    int nx = dtr + 32;
    int M = Bn * L;
    lin(cur, F(base),                nullptr, Ub, M, di, D, Dp, dip, ACT_NONE, 1);
    lin(cur, F(base) + (long)di * D, nullptr, Zb, M, di, D, Dp, dip, ACT_NONE, 1);
    long t1 = (long)M * dip;
    hipLaunchKernelGGL(k_dwconv_silu, dim3((int)((t1 + 255) / 256)), dim3(256), 0, stream,
                       Ub, F(base + 1), F(base + 2), UCb, Bn, L, di, dip);
    lin(UCb, F(base + 3), nullptr,     XDb, M, nx, di,  dip, nxp, ACT_NONE, 1);
    lin(XDb, F(base + 4), F(base + 5), DTb, M, di, dtr, nxp, dip, ACT_SOFTPLUS, 1);
    int st = Bn * di * 16;
    hipLaunchKernelGGL(k_scan, dim3((st + 255) / 256), dim3(256), 0, stream,
                       DTb, UCb, XDb, Zb, F(base + 6), F(base + 7), Ub, Bn, L, di, dip, dtr, nxp);
    float* dst = (cur == ACTA) ? ACTB : ACTA;
    lin(Ub, F(base + 8), nullptr, dst, M, D, di, dip, Dp, ACT_NONE, 1);
    cur = dst;
  };

  up_stage(21, 22, 23, 24, 392, 196, 4);   mamba_stage(25, 196, 512);
  up_stage(34, 35, 36, 37, 196,  98, 8);   mamba_stage(38,  98, 4096);
  up_stage(47, 48, 49, 50,  98,  64, 16);  mamba_stage(51,  64, 32768);

  // ---- final 1x1x1 conv 64 -> 1 straight into d_out (NCDHW since C=1) ----
  lin(cur, F(60), F(61), (float*)d_out, Bn * 32768, 1, 64, 64, 1, ACT_NONE, 0);
}
